// LSTM_24970939859314
// MI455X (gfx1250) — matrix operand order repaired and checked by reading, not compiled, not run on hardware
//
#include <hip/hip_runtime.h>
#include <hip/hip_bf16.h>

// ---------------------------------------------------------------------------
// LSTM on MI455X (gfx1250, wave32, WMMA).
//  VOCAB=50000 EMB=300 HID=128 SEQ=512 BATCH=64, 4H=512.
//  Phase 0: pack w_ih -> bf16 [512][320] (zero-pad K), w_hh -> bf16 [512][128]
//  Phase 1: x_proj[s][b][n] = embeds[s] @ w_ih^T + (b_ih+b_hh)   (WMMA bf16)
//  Phase 2: single-workgroup 512-step scan, w_hh fragments in VGPRs,
//           h bounced via LDS bf16, c in VGPRs, then decode to [64,2].
//  Round 2 change: gate nonlinearities via CDNA5 V_TANH_F32 when available
//  (sigmoid(x) = 0.5*tanh(x/2)+0.5) -> ~3x shorter elementwise chain on the
//  recurrence critical path. Guarded by __has_builtin.
// ---------------------------------------------------------------------------

typedef __bf16 bf16_t;
typedef __bf16 v16bf __attribute__((ext_vector_type(16)));
typedef float  v8f   __attribute__((ext_vector_type(8)));
typedef unsigned int v4u __attribute__((ext_vector_type(4)));

#define SEQ   512
#define BATCH 64
#define EMB   300
#define EMBP  320      // padded K for the input projection
#define HID   128
#define G4H   512      // 4*HID

union Frag { v16bf v; v4u q[2]; };

// A-matrix 16x32 bf16 fragment from row-major [row][ldk] bf16:
//   lane: m = L%16, half = L/16; VGPR0-3: k = kk*32 + half*8 + (0..7)
//                               VGPR4-7: k = kk*32 + 16 + half*8 + (0..7)
__device__ __forceinline__ v16bf load_fragA(const bf16_t* base, int row, int ldk,
                                            int kk, int hlf) {
  const bf16_t* p = base + row * ldk + kk * 32 + hlf * 8;
  Frag f;
  f.q[0] = *reinterpret_cast<const v4u*>(p);
  f.q[1] = *reinterpret_cast<const v4u*>(p + 16);
  return f.v;
}

// B-matrix 32x16 bf16 fragment from row-major [n][ldk] bf16 (i.e. W[n][k]):
//   lane: n = L%16, half = L/16; VGPR0-3: k = kk*32 + half*8 + (0..7), VGPR4-7: k = kk*32 + 16 + half*8 + (0..7).
__device__ __forceinline__ v16bf load_fragB(const bf16_t* base, int row, int ldk,
                                            int kk, int hlf) {
  const bf16_t* p = base + row * ldk + kk * 32 + hlf * 8;
  Frag f;
  f.q[0] = *reinterpret_cast<const v4u*>(p);
  f.q[1] = *reinterpret_cast<const v4u*>(p + 16);
  return f.v;
}

__device__ __forceinline__ v8f wmma_bf16(v16bf a, v16bf b, v8f c) {
  return __builtin_amdgcn_wmma_f32_16x16x32_bf16(false, a, false, b,
                                                 (short)0, c, false, false);
}

// ---- gate nonlinearities -------------------------------------------------
#if __has_builtin(__builtin_amdgcn_tanhf)
__device__ __forceinline__ float tanh_f(float x) {
  return __builtin_amdgcn_tanhf(x);            // v_tanh_f32 (CDNA5 TRANS op)
}
__device__ __forceinline__ float sigmoid_f(float x) {
  return __builtin_fmaf(__builtin_amdgcn_tanhf(0.5f * x), 0.5f, 0.5f);
}
#else
__device__ __forceinline__ float tanh_f(float x) {
  float t = __expf(-2.0f * fabsf(x));          // in (0,1]; overflow-safe
  float r = (1.0f - t) / (1.0f + t);
  return copysignf(r, x);
}
__device__ __forceinline__ float sigmoid_f(float x) {
  return 1.0f / (1.0f + __expf(-x));           // v_exp_f32 + v_rcp_f32
}
#endif

// ---------------------------------------------------------------------------
// Kernel 0: f32 -> bf16 weight packing. w_ih [512][300] -> [512][320] padded,
// w_hh [512][128] -> [512][128].
// ---------------------------------------------------------------------------
__global__ __launch_bounds__(256) void pack_weights(const float* __restrict__ w_ih,
                                                    const float* __restrict__ w_hh,
                                                    bf16_t* __restrict__ wihp,
                                                    bf16_t* __restrict__ whhp) {
  int i = blockIdx.x * blockDim.x + threadIdx.x;
  const int n_ih = G4H * EMBP;                 // 163840
  if (i < n_ih) {
    int g = i / EMBP, e = i % EMBP;
    float v = (e < EMB) ? w_ih[g * EMB + e] : 0.0f;
    wihp[i] = (bf16_t)v;
  }
  int j = i - n_ih;
  if (j >= 0 && j < G4H * HID) {
    whhp[j] = (bf16_t)w_hh[j];
  }
}

// ---------------------------------------------------------------------------
// Kernel 1: x_proj[s][b][n] = sum_e emb[sentence[s,b]][e] * w_ih[n][e] + bias[n]
// grid = (SEQ, 2) blocks, 256 threads (8 waves).
// wave w: m-tile mt=w&3 (16 batch rows), n-block nb=w>>2 covering 8 n-tiles.
// ---------------------------------------------------------------------------
__global__ __launch_bounds__(256, 1)
void xproj_kernel(const int* __restrict__ sentence, const float* __restrict__ emb,
                  const bf16_t* __restrict__ wihp, const float* __restrict__ b_ih,
                  const float* __restrict__ b_hh, float* __restrict__ xproj) {
  __shared__ bf16_t smA[BATCH * EMBP];         // 40 KB staged gathered embeddings
  const int s = blockIdx.x;
  const int nhalf = blockIdx.y;
  const int t = threadIdx.x;

  // Stage: gather + f32->bf16 convert, zero-pad e in [300,320)
  for (int idx = t; idx < BATCH * EMBP; idx += 256) {
    int r = idx / EMBP, c = idx % EMBP;
    int rowid = sentence[s * BATCH + r];
    float v = (c < EMB) ? emb[(long long)rowid * EMB + c] : 0.0f;
    smA[idx] = (bf16_t)v;
  }
  __syncthreads();

  const int w = t >> 5, lane = t & 31, hlf = lane >> 4, lr = lane & 15;
  const int mt = w & 3;                        // 0..3  (batch tile)
  const int nb = w >> 2;                       // 0..1
  const int ntile0 = nhalf * 16 + nb * 8;      // 8 n-tiles per wave

  v8f acc[8];
#pragma unroll
  for (int i = 0; i < 8; ++i) {
    int n = (ntile0 + i) * 16 + lr;
    float bias = b_ih[n] + b_hh[n];            // same for every row m
#pragma unroll
    for (int v = 0; v < 8; ++v) acc[i][v] = bias;
  }

#pragma unroll
  for (int kk = 0; kk < EMBP / 32; ++kk) {     // 10 K-steps
    v16bf a = load_fragA(smA, mt * 16 + lr, EMBP, kk, hlf);
#pragma unroll
    for (int i = 0; i < 8; ++i) {
      v16bf b = load_fragB(wihp, (ntile0 + i) * 16 + lr, EMBP, kk, hlf);
      acc[i] = wmma_bf16(a, b, acc[i]);
    }
  }

  float* outp = xproj + (long long)s * BATCH * G4H;
#pragma unroll
  for (int i = 0; i < 8; ++i) {
    int n = (ntile0 + i) * 16 + lr;
#pragma unroll
    for (int v = 0; v < 8; ++v) {
      int b = mt * 16 + hlf * 8 + v;           // D layout: m = v + 8*half
      outp[b * G4H + n] = acc[i][v];
    }
  }
}

// ---------------------------------------------------------------------------
// Kernel 2: sequential LSTM scan + decode. ONE block, 256 threads (8 waves).
// wave jt (0..7) owns H columns [jt*16, jt*16+16) for ALL gates: its four
// output n-tiles are n = g*128 + jt*16 (g = i,f,g,o) so the c/h update is
// wave-local. w_hh B-fragments (4 gates x 4 K-steps = 128 VGPRs) live in
// registers for the whole 512-step loop; h is bounced through 16 KB of LDS.
// ---------------------------------------------------------------------------
__global__ __launch_bounds__(256, 1)
void lstm_scan(const bf16_t* __restrict__ whhp, const float* __restrict__ xproj,
               const float* __restrict__ w_dec, const float* __restrict__ b_dec,
               float* __restrict__ out) {
  __shared__ bf16_t h_sm[BATCH * HID];         // 16 KB, bf16 h for WMMA A
  __shared__ float  hf_sm[BATCH * HID];        // 32 KB, f32 h for decode

  const int t = threadIdx.x, jt = t >> 5;
  const int lane = t & 31, hlf = lane >> 4, lr = lane & 15;

  // Preload w_hh fragments: Bf[g][kk] covers n-tile g*8+jt, k-step kk.
  v16bf Bf[4][4];
#pragma unroll
  for (int g = 0; g < 4; ++g) {
    int n = (g * 8 + jt) * 16 + lr;
#pragma unroll
    for (int kk = 0; kk < 4; ++kk)
      Bf[g][kk] = load_fragB(whhp, n, HID, kk, hlf);
  }

  float c[4][8];                               // [m-tile][v]  c-state slice
  float hnew[4][8];
#pragma unroll
  for (int mt = 0; mt < 4; ++mt)
#pragma unroll
    for (int v = 0; v < 8; ++v) { c[mt][v] = 0.0f; hnew[mt][v] = 0.0f; }
  __syncthreads();

#pragma unroll 1
  for (int s = 0; s < SEQ; ++s) {
    const float* xp = xproj + (long long)s * BATCH * G4H;
    if (s + 1 < SEQ)                           // global_prefetch_b8 of next step
      __builtin_prefetch(xproj + (long long)(s + 1) * BATCH * G4H + t * 128, 0, 1);

#pragma unroll
    for (int mt = 0; mt < 4; ++mt) {
      v8f acc[4];
#pragma unroll
      for (int g = 0; g < 4; ++g) {            // seed with x_proj (has biases)
        int n = (g * 8 + jt) * 16 + lr;
#pragma unroll
        for (int v = 0; v < 8; ++v)
          acc[g][v] = xp[(mt * 16 + hlf * 8 + v) * G4H + n];
      }
      if (s > 0) {                             // h == 0 at s == 0
#pragma unroll
        for (int kk = 0; kk < 4; ++kk) {
          v16bf a = load_fragA(h_sm, mt * 16 + lr, HID, kk, hlf);
#pragma unroll
          for (int g = 0; g < 4; ++g)
            acc[g] = wmma_bf16(a, Bf[g][kk], acc[g]);
        }
      }
#pragma unroll
      for (int v = 0; v < 8; ++v) {            // torch gate order i,f,g,o
        float ig = sigmoid_f(acc[0][v]);
        float fg = sigmoid_f(acc[1][v]);
        float gg = tanh_f(acc[2][v]);
        float og = sigmoid_f(acc[3][v]);
        float cc = fg * c[mt][v] + ig * gg;
        c[mt][v] = cc;
        hnew[mt][v] = og * tanh_f(cc);
      }
    }
    __syncthreads();                           // all reads of old h done
#pragma unroll
    for (int mt = 0; mt < 4; ++mt)
#pragma unroll
      for (int v = 0; v < 8; ++v)
        h_sm[(mt * 16 + hlf * 8 + v) * HID + jt * 16 + lr] = (bf16_t)hnew[mt][v];
    __syncthreads();                           // new h visible to all waves
  }

  // h_last in f32 for the decode
#pragma unroll
  for (int mt = 0; mt < 4; ++mt)
#pragma unroll
    for (int v = 0; v < 8; ++v)
      hf_sm[(mt * 16 + hlf * 8 + v) * HID + jt * 16 + lr] = hnew[mt][v];
  __syncthreads();

  if (t < BATCH * 2) {                         // out[b][d] = h_last . w_dec[d] + b_dec[d]
    int b = t >> 1, d = t & 1;
    float sum = b_dec[d];
    for (int j = 0; j < HID; ++j)
      sum += hf_sm[b * HID + j] * w_dec[d * HID + j];
    out[b * 2 + d] = sum;
  }
}

// ---------------------------------------------------------------------------
// Launch
// ---------------------------------------------------------------------------
extern "C" void kernel_launch(void* const* d_in, const int* in_sizes, int n_in,
                              void* d_out, int out_size, void* d_ws, size_t ws_size,
                              hipStream_t stream) {
  const int*   sentence = (const int*)  d_in[0];   // [512,64] int32
  const float* emb      = (const float*)d_in[1];   // [50000,300]
  const float* w_ih     = (const float*)d_in[2];   // [512,300]
  const float* w_hh     = (const float*)d_in[3];   // [512,128]
  const float* b_ih     = (const float*)d_in[4];   // [512]
  const float* b_hh     = (const float*)d_in[5];   // [512]
  const float* w_dec    = (const float*)d_in[6];   // [2,128]
  const float* b_dec    = (const float*)d_in[7];   // [2]
  float* out = (float*)d_out;                      // [64,2]

  char* ws = (char*)d_ws;
  bf16_t* wihp  = (bf16_t*)(ws);                   // 512*320*2 = 327,680 B
  bf16_t* whhp  = (bf16_t*)(ws + 327680);          // 512*128*2 = 131,072 B
  float*  xproj = (float*) (ws + 458752);          // 512*64*512*4 = 64 MiB

  pack_weights<<<896, 256, 0, stream>>>(w_ih, w_hh, wihp, whhp);
  xproj_kernel<<<dim3(SEQ, 2), 256, 0, stream>>>(sentence, emb, wihp, b_ih, b_hh, xproj);
  lstm_scan<<<1, 256, 0, stream>>>(whhp, xproj, w_dec, b_dec, out);
}
